// Net_29721173688859
// MI455X (gfx1250) — compile-verified, matrix operand order checked by reading, not run on hardware
//
#include <hip/hip_runtime.h>
#include <hip/hip_bf16.h>

#define NN     8192
#define EE     65536
#define EPAIR  (EE + NN)        // edges + self loops
#define FIN    3247
#define FINP   3264             // roundup32
#define FEXP   7993
#define FEXPP  8000             // roundup32
#define HH     3
#define CC     512
#define HCH    1536
#define NEG_SLOPE 0.2f

typedef __attribute__((ext_vector_type(16))) __bf16 v16bf;
typedef __attribute__((ext_vector_type(8)))  __bf16 v8bf;
typedef __attribute__((ext_vector_type(8)))  float  v8f;
typedef int v4i __attribute__((vector_size(16)));
typedef v4i __attribute__((address_space(3))) * lds_v4i_p;

#if __has_builtin(__builtin_amdgcn_global_load_async_to_lds_b128) && \
    __has_builtin(__builtin_amdgcn_s_wait_asynccnt)
#define HAVE_ASYNC_LDS 1
#define GPTR(p) ((v4i __attribute__((address_space(1))) *)(p))
#define LPTR(p) ((lds_v4i_p)(p))
#else
#define HAVE_ASYNC_LDS 0
#endif

// ------------------------------------------------------------------
// Tiled GEMM: C[M,N] = act(A[M,Kp] @ Bt[N,Kp]^T + bias[N])
// A, Bt pre-converted bf16, Kp % 32 == 0 (zero padded), M % 128 == 0.
// Double-buffered LDS, pipeline unrolled x2 (literal buffer indices);
// staging via global_load_async_to_lds_b128 (ASYNCcnt) when available,
// else b128 load/ds_store fallback.
// Block = 256 threads = 8 waves; block tile 128x64; wave tile 32x32.
// ------------------------------------------------------------------
#define TM 128
#define TN 64
#define TK 32
#define LDA 40   // bf16 elems; 80B row stride (16B aligned, bank-spread)
#define LDB 40

__global__ __launch_bounds__(256)
void gemm_bf16_wmma(const __bf16* __restrict__ A,   // [M, Kp]
                    const __bf16* __restrict__ Bt,  // [N, Kp]
                    const float*  __restrict__ bias,
                    void*         __restrict__ Cout,
                    int M, int K, int N, int ldc, int relu, int out_bf16)
{
    __shared__ __align__(16) __bf16 As[2][TM * LDA];
    __shared__ __align__(16) __bf16 Bs[2][TN * LDB];

    const int tid  = threadIdx.x;
    const int lane = tid & 31;
    const int wave = tid >> 5;
    const int wm   = wave >> 1;        // 0..3 : wave row of 32
    const int wn   = wave & 1;         // 0..1 : wave col of 32
    const int g    = lane >> 4;        // half-wave group
    const int ml   = lane & 15;

    const int rowBase = blockIdx.y * TM;
    const int colBase = blockIdx.x * TN;

    // staging assignments (all b128 transfers, branch-free)
    const int ar = tid >> 1, ah = tid & 1;      // A: row 0..127, 16-col half
    const int bn = tid >> 2, bq = tid & 3;      // B: row 0..63,  8-col quarter
    const int agr = rowBase + ar;               // M % TM == 0 -> always valid
    const int bgn = colBase + bn;
    const int bcl = bgn < N ? bgn : N - 1;      // clamp: junk rows only feed
                                                // cols >= N, zeroed in epilogue
    const int aoff = ar * LDA + 16 * ah;
    const int boff = bn * LDB + 8 * bq;
    const __bf16* aBase = &A[(size_t)agr * K + 16 * ah];
    const __bf16* bBase = &Bt[(size_t)bcl * K + 8 * bq];

#if HAVE_ASYNC_LDS
    // hoisted AS(3) destinations: addrspacecast null-checks become loop-invariant
    lds_v4i_p aL[2][2] = { { LPTR(&As[0][aoff]), LPTR(&As[0][aoff + 8]) },
                           { LPTR(&As[1][aoff]), LPTR(&As[1][aoff + 8]) } };
    lds_v4i_p bL[2]    = {   LPTR(&Bs[0][boff]), LPTR(&Bs[1][boff]) };
#endif

    auto stage = [&](int buf, int k0) {   // buf is always a literal 0/1
#if HAVE_ASYNC_LDS
        __builtin_amdgcn_global_load_async_to_lds_b128(GPTR(aBase + k0),     aL[buf][0], 0, 0);
        __builtin_amdgcn_global_load_async_to_lds_b128(GPTR(aBase + k0 + 8), aL[buf][1], 0, 0);
        __builtin_amdgcn_global_load_async_to_lds_b128(GPTR(bBase + k0),     bL[buf],    0, 0);
#else
        v8bf a0 = *reinterpret_cast<const v8bf*>(aBase + k0);
        v8bf a1 = *reinterpret_cast<const v8bf*>(aBase + k0 + 8);
        v8bf b0 = *reinterpret_cast<const v8bf*>(bBase + k0);
        *reinterpret_cast<v8bf*>(&As[buf][aoff])     = a0;
        *reinterpret_cast<v8bf*>(&As[buf][aoff + 8]) = a1;
        *reinterpret_cast<v8bf*>(&Bs[buf][boff])     = b0;
        if (k0 + TK < K) {
            __builtin_prefetch(aBase + k0 + TK, 0, 1);
            __builtin_prefetch(bBase + k0 + TK, 0, 1);
        }
#endif
    };

    const v8f vzero = {0.f,0.f,0.f,0.f,0.f,0.f,0.f,0.f};
    v8f acc[2][2];
    for (int i = 0; i < 2; ++i)
        for (int j = 0; j < 2; ++j)
            acc[i][j] = vzero;

    auto computeTile = [&](int buf) {     // buf is always a literal 0/1
        // fragment assembly per 16-bit A/B VGPR layout:
        // lane half g: elems 0..7 -> K = 8g..8g+7, elems 8..15 -> K = 16+8g..
        v16bf afrag[2], bfrag[2];
        #pragma unroll
        for (int i = 0; i < 2; ++i) {
            int row = wm * 32 + i * 16 + ml;
            v8bf lo = *reinterpret_cast<const v8bf*>(&As[buf][row * LDA + 8 * g]);
            v8bf hi = *reinterpret_cast<const v8bf*>(&As[buf][row * LDA + 16 + 8 * g]);
            #pragma unroll
            for (int tt = 0; tt < 8; ++tt) { afrag[i][tt] = lo[tt]; afrag[i][8 + tt] = hi[tt]; }
        }
        #pragma unroll
        for (int j = 0; j < 2; ++j) {
            int nrow = wn * 32 + j * 16 + ml;
            v8bf lo = *reinterpret_cast<const v8bf*>(&Bs[buf][nrow * LDB + 8 * g]);
            v8bf hi = *reinterpret_cast<const v8bf*>(&Bs[buf][nrow * LDB + 16 + 8 * g]);
            #pragma unroll
            for (int tt = 0; tt < 8; ++tt) { bfrag[j][tt] = lo[tt]; bfrag[j][8 + tt] = hi[tt]; }
        }
        #pragma unroll
        for (int i = 0; i < 2; ++i)
            #pragma unroll
            for (int j = 0; j < 2; ++j)
                acc[i][j] = __builtin_amdgcn_wmma_f32_16x16x32_bf16(
                    false, afrag[i], false, bfrag[j],
                    (short)0, acc[i][j], false, false);
    };

    const int nk = K / TK;              // K % TK == 0 by construction
    stage(0, 0);

    int t = 0;
    while (t < nk) {
        // ---- tile t lives in buffer 0 ----
        if (t + 1 < nk) {
            stage(1, (t + 1) * TK);
#if HAVE_ASYNC_LDS
            __builtin_amdgcn_s_wait_asynccnt(3);   // tile t retired, t+1 in flight
        } else {
            __builtin_amdgcn_s_wait_asynccnt(0);
#endif
        }
        __syncthreads();
        computeTile(0);
        __syncthreads();
        if (++t >= nk) break;

        // ---- tile t lives in buffer 1 ----
        if (t + 1 < nk) {
            stage(0, (t + 1) * TK);
#if HAVE_ASYNC_LDS
            __builtin_amdgcn_s_wait_asynccnt(3);
        } else {
            __builtin_amdgcn_s_wait_asynccnt(0);
#endif
        }
        __syncthreads();
        computeTile(1);
        __syncthreads();
        ++t;
    }

    // epilogue: C/D layout -> lane half g holds M = r + 8g, N = ml.
    // writes cols [0, ldc): cols >= N get zeros (keeps padded bf16 layouts clean)
    #pragma unroll
    for (int i = 0; i < 2; ++i) {
        #pragma unroll
        for (int j = 0; j < 2; ++j) {
            int gcol = colBase + wn * 32 + j * 16 + ml;
            if (gcol < ldc) {
                float bv = (gcol < N) ? bias[gcol] : 0.f;
                #pragma unroll
                for (int r = 0; r < 8; ++r) {
                    int grow = rowBase + wm * 32 + i * 16 + 8 * g + r;
                    if (grow < M) {
                        float v = (gcol < N) ? acc[i][j][r] + bv : 0.f;
                        if (relu && v < 0.f) v = 0.f;
                        if (out_bf16)
                            ((__bf16*)Cout)[(size_t)grow * ldc + gcol] = (__bf16)v;
                        else
                            ((float*)Cout)[(size_t)grow * ldc + gcol] = v;
                    }
                }
            }
        }
    }
}

// ------------------------------------------------------------------
// Layout conversion kernels (run once per launch; bandwidth-trivial)
// ------------------------------------------------------------------
// f32 [rows,K] -> bf16 [rows,Kp], zero pad K..Kp
__global__ void cvt_pad(const float* __restrict__ in, __bf16* __restrict__ out,
                        int K, int Kp)
{
    int c = blockIdx.x * 256 + threadIdx.x;
    int r = blockIdx.y;
    if (c >= Kp) return;
    out[(size_t)r * Kp + c] = (c < K) ? (__bf16)in[(size_t)r * K + c] : (__bf16)0.f;
}

// f32 [K,N] -> bf16 [N,Kp] (transpose + convert + zero pad)
__global__ __launch_bounds__(256)
void transpose_cvt(const float* __restrict__ in, __bf16* __restrict__ out,
                   int K, int N, int Kp)
{
    __shared__ float tile[32][33];
    int tx = threadIdx.x & 31, ty = threadIdx.x >> 5;
    int kbase = blockIdx.y * 32, nbase = blockIdx.x * 32;
    #pragma unroll
    for (int j = 0; j < 32; j += 8) {
        int k = kbase + ty + j, n = nbase + tx;
        tile[ty + j][tx] = (k < K && n < N) ? in[(size_t)k * N + n] : 0.f;
    }
    __syncthreads();
    #pragma unroll
    for (int j = 0; j < 32; j += 8) {
        int n = nbase + ty + j, k = kbase + tx;
        if (n < N && k < Kp) out[(size_t)n * Kp + k] = (__bf16)tile[tx][ty + j];
    }
}

// ------------------------------------------------------------------
// Segment-softmax attention (GATv2) helper kernels
// ------------------------------------------------------------------
__device__ __forceinline__ unsigned fkey(float f) {
    unsigned x = __float_as_uint(f);
    return (x & 0x80000000u) ? ~x : (x | 0x80000000u);   // monotone float->uint
}
__device__ __forceinline__ float kval(unsigned k) {
    unsigned x = (k & 0x80000000u) ? (k ^ 0x80000000u) : ~k;
    return __uint_as_float(x);
}

__global__ void seg_init(unsigned* __restrict__ mk, float* __restrict__ z,
                         float* __restrict__ agg)
{
    size_t i = (size_t)blockIdx.x * 256 + threadIdx.x;
    if (i < (size_t)NN * HH) { mk[i] = 0u; z[i] = 0.f; }
    if (i < (size_t)NN * HCH) agg[i] = 0.f;
}

// one wave per (edge, head): logit = a . leakyrelu(xl[src]+xr[dst]); seg-max
__global__ __launch_bounds__(256)
void edge_logit(const int* __restrict__ ei, const float* __restrict__ xl,
                const float* __restrict__ xr, const float* __restrict__ att,
                float* __restrict__ logit, unsigned* __restrict__ mk)
{
    int wid  = blockIdx.x * 8 + (threadIdx.x >> 5);
    int lane = threadIdx.x & 31;
    if (wid >= EPAIR * HH) return;
    int e = wid / HH, h = wid % HH;
    int src = (e < EE) ? ei[e]      : (e - EE);
    int dst = (e < EE) ? ei[EE + e] : (e - EE);
    const float* pl = xl  + (size_t)src * HCH + h * CC;
    const float* pr = xr  + (size_t)dst * HCH + h * CC;
    const float* pa = att + h * CC;
    float s = 0.f;
    for (int c = lane; c < CC; c += 32) {
        float v = pl[c] + pr[c];
        v = v > 0.f ? v : NEG_SLOPE * v;
        s += v * pa[c];
    }
    #pragma unroll
    for (int off = 16; off; off >>= 1) s += __shfl_xor(s, off, 32);
    if (lane == 0) {
        logit[wid] = s;
        atomicMax(&mk[dst * HH + h], fkey(s));
    }
}

__global__ void edge_expsum(const int* __restrict__ ei, float* __restrict__ pbuf,
                            const unsigned* __restrict__ mk, float* __restrict__ z)
{
    int i = blockIdx.x * 256 + threadIdx.x;
    if (i >= EPAIR * HH) return;
    int e = i / HH, h = i % HH;
    int dst = (e < EE) ? ei[EE + e] : (e - EE);
    float p = __expf(pbuf[i] - kval(mk[dst * HH + h]));
    pbuf[i] = p;
    atomicAdd(&z[dst * HH + h], p);
}

// one wave per (edge, head): agg[dst] += alpha * xl[src]
__global__ __launch_bounds__(256)
void edge_scatter(const int* __restrict__ ei, const float* __restrict__ pbuf,
                  const float* __restrict__ z, const float* __restrict__ xl,
                  float* __restrict__ agg)
{
    int wid  = blockIdx.x * 8 + (threadIdx.x >> 5);
    int lane = threadIdx.x & 31;
    if (wid >= EPAIR * HH) return;
    int e = wid / HH, h = wid % HH;
    int src = (e < EE) ? ei[e]      : (e - EE);
    int dst = (e < EE) ? ei[EE + e] : (e - EE);
    float alpha = pbuf[wid] / z[dst * HH + h];
    const float* pl = xl  + (size_t)src * HCH + h * CC;
    float*       po = agg + (size_t)dst * HCH + h * CC;
    for (int c = lane; c < CC; c += 32)
        atomicAdd(&po[c], alpha * pl[c]);
}

// out_bf16 = relu(agg + bias): feeds next layer's GEMM directly
__global__ void bias_act(const float* __restrict__ agg, const float* __restrict__ ob,
                         __bf16* __restrict__ out)
{
    size_t i = (size_t)blockIdx.x * 256 + threadIdx.x;
    if (i >= (size_t)NN * HCH) return;
    int col = (int)(i % HCH);
    float v = agg[i] + ob[col];
    out[i] = (__bf16)(v > 0.f ? v : 0.f);
}

__global__ void concat_kernel(const __bf16* __restrict__ h, const __bf16* __restrict__ cell,
                              __bf16* __restrict__ out)
{
    size_t i = (size_t)blockIdx.x * 256 + threadIdx.x;
    if (i >= (size_t)NN * 2048) return;
    int row = (int)(i / 2048), col = (int)(i % 2048);
    out[i] = (col < HCH) ? h[(size_t)row * HCH + col]
                         : cell[(size_t)row * 512 + (col - HCH)];
}

// ------------------------------------------------------------------
extern "C" void kernel_launch(void* const* d_in, const int* in_sizes, int n_in,
                              void* d_out, int out_size, void* d_ws, size_t ws_size,
                              hipStream_t stream)
{
    (void)in_sizes; (void)n_in; (void)out_size; (void)ws_size;
    const float* x    = (const float*)d_in[0];
    const int*   ei   = (const int*)  d_in[1];
    const float* expm = (const float*)d_in[2];
    const float* w1l = (const float*)d_in[3];  const float* b1l = (const float*)d_in[4];
    const float* w1r = (const float*)d_in[5];  const float* b1r = (const float*)d_in[6];
    const float* a1  = (const float*)d_in[7];  const float* o1  = (const float*)d_in[8];
    const float* w2l = (const float*)d_in[9];  const float* b2l = (const float*)d_in[10];
    const float* w2r = (const float*)d_in[11]; const float* b2r = (const float*)d_in[12];
    const float* a2  = (const float*)d_in[13]; const float* o2  = (const float*)d_in[14];
    const float* w3l = (const float*)d_in[15]; const float* b3l = (const float*)d_in[16];
    const float* w3r = (const float*)d_in[17]; const float* b3r = (const float*)d_in[18];
    const float* a3  = (const float*)d_in[19]; const float* o3  = (const float*)d_in[20];
    const float* e1w = (const float*)d_in[21]; const float* e1b = (const float*)d_in[22];
    const float* e2w = (const float*)d_in[23]; const float* e2b = (const float*)d_in[24];
    const float* e3w = (const float*)d_in[25]; const float* e3b = (const float*)d_in[26];
    const float* l1w = (const float*)d_in[27]; const float* l1b = (const float*)d_in[28];
    const float* l2w = (const float*)d_in[29]; const float* l2b = (const float*)d_in[30];
    const float* l3w = (const float*)d_in[31]; const float* l3b = (const float*)d_in[32];
    float* out = (float*)d_out;

    // ---------------- workspace carve-out (byte based) ----------------
    char* base = (char*)d_ws;
    size_t off = 0;
    auto alloc = [&](size_t bytes) -> char* {
        char* p = base + off; off += (bytes + 255) & ~(size_t)255; return p;
    };
    // f32 block: xl | xr | agg  (also hosts exp_bf alias after GNN trunk)
    char* f32blk = alloc((size_t)NN * HCH * 4 * 3);
    float* xl  = (float*)f32blk;
    float* xr  = xl + (size_t)NN * HCH;
    float* agg = xr + (size_t)NN * HCH;
    __bf16* exp_bf = (__bf16*)f32blk;           // [NN, FEXPP] alias, 131MB <= 151MB

    float*    pbuf = (float*)   alloc((size_t)EPAIR * HH * 4);
    unsigned* mk   = (unsigned*)alloc((size_t)NN * HH * 4);
    float*    z    = (float*)   alloc((size_t)NN * HH * 4);

    // x_bf region, later reused for concat + head1 activations
    char* xbf_blk = alloc((size_t)NN * FINP * 2);
    __bf16* x_bf   = (__bf16*)xbf_blk;                               // [NN, FINP]
    __bf16* cat_bf = (__bf16*)xbf_blk;                               // [NN, 2048]
    __bf16* hd1_bf = (__bf16*)(xbf_blk + (size_t)NN * 2048 * 2);     // [NN, 1024]

    __bf16* h_bf    = (__bf16*)alloc((size_t)NN * HCH  * 2);  // GAT output (bf16)
    __bf16* big_bf  = (__bf16*)alloc((size_t)NN * 4000 * 2);
    __bf16* mid_bf  = (__bf16*)alloc((size_t)NN * 1504 * 2);
    __bf16* cell_bf = (__bf16*)alloc((size_t)NN * 512  * 2);
    __bf16* hd2_bf  = (__bf16*)alloc((size_t)NN * 512  * 2);

    // transposed bf16 weights [N, Kp] — ~123MB total, L2-resident on MI455X
    __bf16* w1lT = (__bf16*)alloc((size_t)HCH  * FINP  * 2);
    __bf16* w1rT = (__bf16*)alloc((size_t)HCH  * FINP  * 2);
    __bf16* w2lT = (__bf16*)alloc((size_t)HCH  * HCH   * 2);
    __bf16* w2rT = (__bf16*)alloc((size_t)HCH  * HCH   * 2);
    __bf16* w3lT = (__bf16*)alloc((size_t)HCH  * HCH   * 2);
    __bf16* w3rT = (__bf16*)alloc((size_t)HCH  * HCH   * 2);
    __bf16* e1T  = (__bf16*)alloc((size_t)4000 * FEXPP * 2);
    __bf16* e2T  = (__bf16*)alloc((size_t)1500 * 4000  * 2);
    __bf16* e3T  = (__bf16*)alloc((size_t)512  * 1504  * 2);
    __bf16* l1T  = (__bf16*)alloc((size_t)1024 * 2048  * 2);
    __bf16* l2T  = (__bf16*)alloc((size_t)512  * 1024  * 2);
    __bf16* l3T  = (__bf16*)alloc((size_t)1    * 512   * 2);

    dim3 blk(256);
    auto gemm = [&](const __bf16* A, const __bf16* Bt, const float* b, void* Cm,
                    int M, int Kp, int N, int ldc, int relu, int obf) {
        dim3 grid((N + TN - 1) / TN, (M + TM - 1) / TM);
        gemm_bf16_wmma<<<grid, blk, 0, stream>>>(A, Bt, b, Cm, M, Kp, N, ldc, relu, obf);
    };
    auto wT = [&](const float* w, __bf16* wt, int K, int N, int Kp) {
        dim3 grid((N + 31) / 32, (Kp + 31) / 32);
        transpose_cvt<<<grid, blk, 0, stream>>>(w, wt, K, N, Kp);
    };
    const int gi_nhc  = (int)(((size_t)NN * HCH + 255) / 256);
    const int gi_pair = (EPAIR * HH + 255) / 256;
    const int gi_wave = (EPAIR * HH) / 8;        // 221184 / 8 = 27648 exact

    auto gat = [&](const __bf16* hsrc, int Kp,
                   const __bf16* wlT, const float* bl,
                   const __bf16* wrT, const float* br,
                   const float* att, const float* ob) {
        gemm(hsrc, wlT, bl, xl, NN, Kp, HCH, HCH, 0, 0);
        gemm(hsrc, wrT, br, xr, NN, Kp, HCH, HCH, 0, 0);
        seg_init    <<<gi_nhc,  blk, 0, stream>>>(mk, z, agg);
        edge_logit  <<<gi_wave, blk, 0, stream>>>(ei, xl, xr, att, pbuf, mk);
        edge_expsum <<<gi_pair, blk, 0, stream>>>(ei, pbuf, mk, z);
        edge_scatter<<<gi_wave, blk, 0, stream>>>(ei, pbuf, z, xl, agg);
        bias_act    <<<gi_nhc,  blk, 0, stream>>>(agg, ob, h_bf);
    };

    // ---- one-time layout conversions ----
    cvt_pad<<<dim3((FINP + 255) / 256, NN), blk, 0, stream>>>(x, x_bf, FIN, FINP);
    wT(w1l, w1lT, FIN,  HCH,  FINP);
    wT(w1r, w1rT, FIN,  HCH,  FINP);
    wT(w2l, w2lT, HCH,  HCH,  HCH);
    wT(w2r, w2rT, HCH,  HCH,  HCH);
    wT(w3l, w3lT, HCH,  HCH,  HCH);
    wT(w3r, w3rT, HCH,  HCH,  HCH);
    wT(e1w, e1T,  FEXP, 4000, FEXPP);
    wT(e2w, e2T,  4000, 1500, 4000);
    wT(e3w, e3T,  1500, 512,  1504);
    wT(l1w, l1T,  2048, 1024, 2048);
    wT(l2w, l2T,  1024, 512,  1024);
    wT(l3w, l3T,  512,  1,    512);

    // ---- GNN trunk (x_bf lifetime ends after layer 1 GEMMs) ----
    gat(x_bf, FINP, w1lT, b1l, w1rT, b1r, a1, o1);
    gat(h_bf, HCH,  w2lT, b2l, w2rT, b2r, a2, o2);
    gat(h_bf, HCH,  w3lT, b3l, w3rT, b3r, a3, o3);

    // ---- expression encoder (exp_bf aliases xl/xr/agg, now dead) ----
    cvt_pad<<<dim3((FEXPP + 255) / 256, NN), blk, 0, stream>>>(expm, exp_bf, FEXP, FEXPP);
    gemm(exp_bf, e1T, e1b, big_bf,  NN, FEXPP, 4000, 4000, 1, 1);
    gemm(big_bf, e2T, e2b, mid_bf,  NN, 4000,  1500, 1504, 1, 1);
    gemm(mid_bf, e3T, e3b, cell_bf, NN, 1504,  512,  512,  0, 1);

    // ---- concat + head MLP (cat/hd1 alias x_bf region, now dead) ----
    concat_kernel<<<(int)(((size_t)NN * 2048 + 255) / 256), blk, 0, stream>>>(h_bf, cell_bf, cat_bf);
    gemm(cat_bf, l1T, l1b, hd1_bf, NN, 2048, 1024, 1024, 1, 1);
    gemm(hd1_bf, l2T, l2b, hd2_bf, NN, 1024, 512,  512,  1, 1);
    gemm(hd2_bf, l3T, l3b, out,    NN, 512,  1,    1,    0, 0);
}
